// MoE_78039555768543
// MI455X (gfx1250) — matrix operand order repaired and checked by reading, not compiled, not run on hardware
//
#include <hip/hip_runtime.h>
#include <hip/hip_bf16.h>

typedef __attribute__((ext_vector_type(16))) __bf16        v16bf;
typedef __attribute__((ext_vector_type(8)))  float         v8f;
typedef __attribute__((ext_vector_type(4)))  unsigned int  u32x4;
typedef __attribute__((ext_vector_type(8)))  unsigned int  u32x8;

#define NTOK  8192      // B*S tokens
#define DDIM  2048
#define NEXP  8
#define FDIM  2048
#define MTILE 64
#define ASTR  40        // A panel row stride in bf16 (bank-conflict-safe, 16B aligned)
#define NTILES 264      // ceil((2*NTOK + NEXP*(MTILE-1)) / MTILE)

__device__ __forceinline__ unsigned short f2bf(float f) {
    unsigned u = __builtin_bit_cast(unsigned, f);
    u += 0x7fffu + ((u >> 16) & 1u);        // round-to-nearest-even
    return (unsigned short)(u >> 16);
}

__device__ __forceinline__ v16bf pack16(u32x4 lo, u32x4 hi) {
    return __builtin_bit_cast(v16bf, __builtin_shufflevector(lo, hi, 0, 1, 2, 3, 4, 5, 6, 7));
}

// ---------------- x: f32 -> bf16 ----------------
__global__ __launch_bounds__(256) void k_cvt_x(const float* __restrict__ x,
                                               unsigned short* __restrict__ xb) {
    const size_t i = ((size_t)blockIdx.x * 256 + threadIdx.x) * 4;
    const float4 v = *(const float4*)(x + i);
    uint2 p;
    p.x = (unsigned)f2bf(v.x) | ((unsigned)f2bf(v.y) << 16);
    p.y = (unsigned)f2bf(v.z) | ((unsigned)f2bf(v.w) << 16);
    *(uint2*)(xb + i) = p;
}

// ---------------- We[e][d][f] f32 -> WbT[e][f][d] bf16 (transpose) ----------------
__global__ __launch_bounds__(256) void k_cvt_wt(const float* __restrict__ we,
                                                unsigned short* __restrict__ wt) {
    __shared__ float tile[32][33];
    const int tx = threadIdx.x, ty = threadIdx.y;
    const int f0 = blockIdx.x * 32, d0 = blockIdx.y * 32, e = blockIdx.z;
    const size_t ebase = (size_t)e * DDIM * FDIM;
#pragma unroll
    for (int rr = 0; rr < 4; ++rr) {
        const int d = ty + rr * 8;
        tile[d][tx] = we[ebase + (size_t)(d0 + d) * FDIM + f0 + tx];
    }
    __syncthreads();
    const size_t obase = (size_t)e * FDIM * DDIM;
#pragma unroll
    for (int rr = 0; rr < 4; ++rr) {
        const int f = ty + rr * 8;
        wt[obase + (size_t)(f0 + f) * DDIM + d0 + tx] = f2bf(tile[tx][f]);
    }
}

// ---------------- router: logits, top-2, softmax ----------------
__global__ __launch_bounds__(256) void k_router(const float* __restrict__ x,
                                                const float* __restrict__ wr,
                                                int* __restrict__ selpk,
                                                float* __restrict__ wpair,
                                                int* __restrict__ cnt) {
    const int lane = threadIdx.x & 31;
    const int wave = threadIdx.x >> 5;
    const int token = blockIdx.x * 8 + wave;
    const float* xrow = x + (size_t)token * DDIM;

    float acc[8];
#pragma unroll
    for (int e = 0; e < 8; ++e) acc[e] = 0.f;

    for (int d = lane; d < DDIM; d += 32) {
        const float xv = xrow[d];
        const float4 w0 = *(const float4*)(wr + d * 8);
        const float4 w1 = *(const float4*)(wr + d * 8 + 4);
        acc[0] += xv * w0.x; acc[1] += xv * w0.y;
        acc[2] += xv * w0.z; acc[3] += xv * w0.w;
        acc[4] += xv * w1.x; acc[5] += xv * w1.y;
        acc[6] += xv * w1.z; acc[7] += xv * w1.w;
    }
#pragma unroll
    for (int e = 0; e < 8; ++e) {
        float v = acc[e];
#pragma unroll
        for (int off = 16; off >= 1; off >>= 1) v += __shfl_xor(v, off, 32);
        acc[e] = v;
    }
    if (lane == 0) {
        int e0 = 0; float l0 = acc[0];
#pragma unroll
        for (int e = 1; e < 8; ++e) if (acc[e] > l0) { l0 = acc[e]; e0 = e; }
        int e1 = (e0 == 0) ? 1 : 0; float l1 = acc[e1];
#pragma unroll
        for (int e = 0; e < 8; ++e)
            if (e != e0 && acc[e] > l1) { l1 = acc[e]; e1 = e; }
        const float t = __expf(l1 - l0);
        const float g0 = 1.f / (1.f + t);
        selpk[token] = e0 | (e1 << 8);
        wpair[2 * token]     = g0;
        wpair[2 * token + 1] = 1.f - g0;
        atomicAdd(&cnt[e0], 1);
        atomicAdd(&cnt[e1], 1);
    }
}

// ---------------- per-expert segment offsets (padded to MTILE) ----------------
__global__ void k_offsets(const int* __restrict__ cnt, int* __restrict__ seg) {
    if (threadIdx.x == 0 && blockIdx.x == 0) {
        int s = 0;
        for (int e = 0; e < NEXP; ++e) {
            seg[e] = s;
            s += (cnt[e] + (MTILE - 1)) & ~(MTILE - 1);
        }
        seg[NEXP] = s;
    }
}

// ---------------- stable compaction: one block per expert ----------------
// Pads segments with (token 0, weight 0): weight 0 exactly nullifies the
// contribution, so the GEMM needs no validity branches anywhere.
__global__ __launch_bounds__(256) void k_compact(const int* __restrict__ selpk,
                                                 const float* __restrict__ wpair,
                                                 const int* __restrict__ seg,
                                                 int* __restrict__ tokens,
                                                 float* __restrict__ wlist) {
    const int e = blockIdx.x;
    const int tid = threadIdx.x;
    const int lane = tid & 31;
    const int wave = tid >> 5;
    __shared__ int s_wtot[8];
    const int s0 = seg[e];
    const int segLen = seg[e + 1] - s0;

    int base = 0;
    for (int c = 0; c < NTOK; c += 256) {
        const int tok = c + tid;
        const int pk = selpk[tok];
        const int e0 = pk & 0xff, e1 = (pk >> 8) & 0xff;
        int flag = 0; float wgt = 0.f;
        if (e0 == e)      { flag = 1; wgt = wpair[2 * tok]; }
        else if (e1 == e) { flag = 1; wgt = wpair[2 * tok + 1]; }
        const unsigned mask = (unsigned)__ballot(flag);
        if (lane == 0) s_wtot[wave] = __popc(mask);
        __syncthreads();
        int off = 0, tot = 0;
#pragma unroll
        for (int w = 0; w < 8; ++w) { if (w < wave) off += s_wtot[w]; tot += s_wtot[w]; }
        if (flag) {
            const int pos = base + off + __popc(mask & ((1u << lane) - 1u));
            tokens[s0 + pos] = tok;
            wlist[s0 + pos]  = wgt;
        }
        base += tot;
        __syncthreads();
    }
    for (int i = base + tid; i < segLen; i += 256) {
        tokens[s0 + i] = 0;      // valid address, zero gate
        wlist[s0 + i]  = 0.f;
    }
}

// ---- one 32-K GEMM step: stage A(next) into `swrite`, prefetch B(next) into
// (nbl,nbh), run 4 WMMAs on A(cur)=aread with B(cur)=(bl,bh). One barrier. ----
__device__ __forceinline__ void gemm_step(const unsigned short* __restrict__ aread,
                                          unsigned short* __restrict__ swrite,
                                          const unsigned short* __restrict__ aptr,
                                          const unsigned short* __restrict__ bptr,
                                          int knext, int kb16,
                                          u32x4& bl, u32x4& bh, v8f (&acc)[4]) {
    // stage next A panel slice (unconditional: pad rows are token 0)
    const u32x4 av = *(const u32x4*)(aptr + knext);
    // prefetch next B operand into registers (hides global latency behind WMMA)
    const u32x4 nbl = *(const u32x4*)(bptr + knext + kb16);
    const u32x4 nbh = *(const u32x4*)(bptr + knext + kb16 + 8);
    *(u32x4*)swrite = av;
    __builtin_prefetch(bptr + knext + 96, 0, 1);

    const v16bf bmat = pack16(bl, bh);
    u32x4 alo[4], ahi[4];
#pragma unroll
    for (int m = 0; m < 4; ++m) {
        alo[m] = *(const u32x4*)(aread + m * 16 * ASTR);
        ahi[m] = *(const u32x4*)(aread + m * 16 * ASTR + 8);
    }
#pragma unroll
    for (int m = 0; m < 4; ++m) {
        acc[m] = __builtin_amdgcn_wmma_f32_16x16x32_bf16(
            false, pack16(alo[m], ahi[m]), false, bmat, (short)0, acc[m], false, false);
    }
    bl = nbl; bh = nbh;
    __syncthreads();
}

// ---------------- grouped expert GEMM: M=64, N=128 per block, bf16 WMMA,
// double-buffered A panel, software-pipelined B ----------------
__global__ __launch_bounds__(256) void k_moe_gemm(const unsigned short* __restrict__ xb,
                                                  const unsigned short* __restrict__ wt,
                                                  const float* __restrict__ be,
                                                  const int* __restrict__ tokens,
                                                  const float* __restrict__ wlist,
                                                  const int* __restrict__ seg,
                                                  float* __restrict__ out) {
    __shared__ int s_seg[NEXP + 1];
    __shared__ int s_tok[MTILE];
    __shared__ float s_twt[MTILE];
    __shared__ __align__(16) unsigned short s_a[2][MTILE * ASTR];

    const int tid = threadIdx.x;
    if (tid < NEXP + 1) s_seg[tid] = seg[tid];
    __syncthreads();

    const int rowBase = blockIdx.x * MTILE;
    if (rowBase >= s_seg[NEXP]) return;

    int e = 0;
#pragma unroll
    for (int i = 0; i < NEXP - 1; ++i) if (rowBase >= s_seg[i + 1]) e = i + 1;

    if (tid < MTILE) {
        s_tok[tid] = tokens[rowBase + tid];
        s_twt[tid] = wlist[rowBase + tid];
    }
    __syncthreads();

    // hoisted, loop-invariant gather pointer for this thread's A staging slice
    const int arow = tid >> 2;   // 0..63
    const int aseg = tid & 3;    // 4 x 16B per 64B row
    const unsigned short* aptr = xb + (size_t)s_tok[arow] * DDIM + aseg * 8;
    unsigned short* sw0 = &s_a[0][arow * ASTR + aseg * 8];
    unsigned short* sw1 = &s_a[1][arow * ASTR + aseg * 8];

    const int lane = tid & 31;
    const int wave = tid >> 5;
    const int nloc = lane & 15;
    const int n = blockIdx.y * 128 + wave * 16 + nloc;
    const int kb8  = (lane < 16) ? 0 : 16;   // A-operand K-half base (contiguous, matches B)
    const int kb16 = (lane < 16) ? 0 : 16;   // B-operand K-half base

    const unsigned short* bptr = wt + ((size_t)e * FDIM + n) * DDIM;
    const float bias = be[e * FDIM + n];
    const unsigned short* ar0 = &s_a[0][nloc * ASTR + kb8];
    const unsigned short* ar1 = &s_a[1][nloc * ASTR + kb8];

    v8f acc[4];
#pragma unroll
    for (int m = 0; m < 4; ++m) acc[m] = (v8f){0.f,0.f,0.f,0.f,0.f,0.f,0.f,0.f};

    // prologue: stage k=0 into buf0, preload B(0)
    *(u32x4*)sw0 = *(const u32x4*)(aptr);
    u32x4 bl = *(const u32x4*)(bptr + kb16);
    u32x4 bh = *(const u32x4*)(bptr + kb16 + 8);
    __syncthreads();

    for (int k0 = 0; k0 < DDIM; k0 += 64) {
        const int kn2 = (k0 + 64 < DDIM) ? (k0 + 64) : (DDIM - 32);
        gemm_step(ar0, sw1, aptr, bptr, k0 + 32, kb16, bl, bh, acc);  // compute k0   (buf0)
        gemm_step(ar1, sw0, aptr, bptr, kn2,     kb16, bl, bh, acc);  // compute k0+32(buf1)
    }

    // gate-scaled scatter accumulate (pads have weight 0 -> add exact zero)
    const int mh = (lane < 16) ? 0 : 8;
#pragma unroll
    for (int m = 0; m < 4; ++m) {
#pragma unroll
        for (int r = 0; r < 8; ++r) {
            const int row = m * 16 + r + mh;
            atomicAdd(out + (size_t)s_tok[row] * FDIM + n,
                      s_twt[row] * (acc[m][r] + bias));
        }
    }
}

extern "C" void kernel_launch(void* const* d_in, const int* in_sizes, int n_in,
                              void* d_out, int out_size, void* d_ws, size_t ws_size,
                              hipStream_t stream) {
    const float* x  = (const float*)d_in[0];   // [4,2048,2048]
    const float* wr = (const float*)d_in[1];   // [2048,8]
    const float* we = (const float*)d_in[2];   // [8,2048,2048]
    const float* be = (const float*)d_in[3];   // [8,2048]
    float* out = (float*)d_out;                // [4,2048,2048]

    char* ws = (char*)d_ws;
    unsigned short* xb     = (unsigned short*)(ws);                 //  33,554,432 B
    unsigned short* wtp    = (unsigned short*)(ws +  33554432);     //  67,108,864 B
    int*            tokens = (int*)           (ws + 100663296);     //  67,584 B
    float*          wlist  = (float*)         (ws + 100730880);     //  67,584 B
    int*            selpk  = (int*)           (ws + 100798464);     //  32,768 B
    float*          wpair  = (float*)         (ws + 100831232);     //  65,536 B
    int*            cnt    = (int*)           (ws + 100896768);     //  32 B
    int*            seg    = (int*)           (ws + 100897024);     //  36 B

    hipMemsetAsync(cnt, 0, 32, stream);
    hipMemsetAsync(out, 0, (size_t)out_size * sizeof(float), stream);

    k_cvt_x   <<<16384, 256, 0, stream>>>(x, xb);
    k_cvt_wt  <<<dim3(FDIM / 32, DDIM / 32, NEXP), dim3(32, 8), 0, stream>>>(we, wtp);
    k_router  <<<NTOK / 8, 256, 0, stream>>>(x, wr, selpk, wpair, cnt);
    k_offsets <<<1, 1, 0, stream>>>(cnt, seg);
    k_compact <<<NEXP, 256, 0, stream>>>(selpk, wpair, seg, tokens, wlist);
    k_moe_gemm<<<dim3(NTILES, FDIM / 128), 256, 0, stream>>>(xb, wtp, be, tokens, wlist,
                                                             seg, out);
}
